// DeeperGCN_31765578121360
// MI455X (gfx1250) — matrix operand order repaired and checked by reading, not compiled, not run on hardware
//
#include <hip/hip_runtime.h>
#include <hip/hip_bf16.h>

// ---- CDNA5 WMMA types ----
typedef __attribute__((ext_vector_type(16))) __bf16 v16bf;
typedef __attribute__((ext_vector_type(8)))  float  v8f;

#define N_NODES 80000
#define N_EDGES 1280000
#define H       64
#define H2      128
#define L_LAYERS 4
#define NGRAPH  256
#define ATOM_V  100
#define ATOM_F  9
#define BOND_V  10
#define BOND_F  3
#define EPSM    1e-7f

// -------------------- utility --------------------
__global__ void k_zero(float* p, long n) {
    long i = (long)blockIdx.x * blockDim.x + threadIdx.x;
    if (i < n) p[i] = 0.f;
}

// -------------------- atom encoder: xn[n,c] = sum_f atom_emb[x[n,f]+f*100][c] ----
__global__ void k_atom_enc(const int* __restrict__ x, const float* __restrict__ aemb,
                           float* __restrict__ xn) {
    long i = (long)blockIdx.x * blockDim.x + threadIdx.x;
    if (i >= (long)N_NODES * H) return;
    int n = (int)(i >> 6), c = (int)(i & 63);
    float s = 0.f;
#pragma unroll
    for (int f = 0; f < ATOM_F; ++f) {
        int row = x[n * ATOM_F + f] + f * ATOM_V;
        s += aemb[(long)row * H + c];
    }
    xn[i] = s;
}

// -------------------- weight pack to bf16 B-fragment layout --------------------
// dst index = ((nt*(K/32)+kc)*32 + lane)*16 + j  maps to  W[k][col], where
//   col = nt*16 + (lane&15),  k = kc*32 + (lane>=16 ? 16 : 0) + j
// (contiguous k order used here, not the ISA table order: lanes 0-15 hold K=0..15, lanes 16-31 K=16..31,
//  2 K-values per VGPR). Each lane's 16 bf16 fragment becomes one contiguous 32B load.
__global__ void k_pack(const float* __restrict__ W, unsigned short* __restrict__ dst,
                       int Kdim, int Ncols) {
    int per = Kdim * Ncols;
    int idx = blockIdx.x * blockDim.x + threadIdx.x;
    if (idx >= per * L_LAYERS) return;
    int l = idx / per;
    int r = idx - l * per;
    int KC = Kdim / 32;
    int j    = r & 15;
    int lane = (r >> 4) & 31;
    int kc   = (r >> 9) % KC;
    int nt   = r / (512 * KC);
    int k   = kc * 32 + ((lane >> 4) << 4) + j;
    int col = nt * 16 + (lane & 15);
    __bf16 b = (__bf16)W[(long)l * per + (long)k * Ncols + col];
    dst[idx] = *(const unsigned short*)&b;
}

// -------------------- per-node LayerNorm + relu (H=64) --------------------
__global__ __launch_bounds__(H) void k_ln_node(const float* __restrict__ in,
                                               float* __restrict__ out,
                                               const float* __restrict__ g,
                                               const float* __restrict__ b) {
    __shared__ float ss[H], sq[H];
    int n = blockIdx.x, c = threadIdx.x;
    float v = in[(long)n * H + c];
    ss[c] = v; sq[c] = v * v;
    __syncthreads();
    for (int s = H / 2; s > 0; s >>= 1) {
        if (c < s) { ss[c] += ss[c + s]; sq[c] += sq[c + s]; }
        __syncthreads();
    }
    float mu  = ss[0] * (1.f / H);
    float var = sq[0] * (1.f / H) - mu * mu;
    float r = (v - mu) * rsqrtf(var + 1e-5f) * g[c] + b[c];
    out[(long)n * H + c] = r > 0.f ? r : 0.f;
}

// -------------------- final LN + relu + per-graph pooling --------------------
__global__ __launch_bounds__(H) void k_ln_pool(const float* __restrict__ in,
                                               const int* __restrict__ batch,
                                               const float* __restrict__ g,
                                               const float* __restrict__ b,
                                               float* __restrict__ psum,
                                               float* __restrict__ pcnt) {
    __shared__ float ss[H], sq[H];
    int n = blockIdx.x, c = threadIdx.x;
    float v = in[(long)n * H + c];
    ss[c] = v; sq[c] = v * v;
    __syncthreads();
    for (int s = H / 2; s > 0; s >>= 1) {
        if (c < s) { ss[c] += ss[c + s]; sq[c] += sq[c + s]; }
        __syncthreads();
    }
    float mu  = ss[0] * (1.f / H);
    float var = sq[0] * (1.f / H) - mu * mu;
    float r = (v - mu) * rsqrtf(var + 1e-5f) * g[c] + b[c];
    r = r > 0.f ? r : 0.f;
    int gi = batch[n];
    atomicAdd(&psum[(long)gi * H + c], r);
    if (c == 0) atomicAdd(&pcnt[gi], 1.f);
}

// -------------------- edge pass 1: per-(dst,channel) max of m --------------------
// m = (relu(h[src]+ea)+EPS)*t >= 0, so uint-bitpattern atomicMax == float max.
// Bond table (7.7KB) staged in LDS; never materialize ea[E,64] (327MB of HBM traffic).
__global__ __launch_bounds__(256) void k_edge_max(const int* __restrict__ ei,
                                                  const int* __restrict__ ea,
                                                  const float* __restrict__ bemb,
                                                  const float* __restrict__ h,
                                                  const float* __restrict__ t, int l,
                                                  unsigned* __restrict__ mx) {
    __shared__ float sbe[BOND_F * BOND_V * H];
    for (int i = threadIdx.x; i < BOND_F * BOND_V * H; i += 256) sbe[i] = bemb[i];
    __syncthreads();
    int e = blockIdx.x * 4 + (threadIdx.x >> 6);
    int c = threadIdx.x & 63;
    if (e >= N_EDGES) return;
    float tl = t[l];
    int src = ei[e], dst = ei[N_EDGES + e];
    float eav = sbe[ea[e * 3 + 0] * H + c]
              + sbe[(ea[e * 3 + 1] + BOND_V) * H + c]
              + sbe[(ea[e * 3 + 2] + 2 * BOND_V) * H + c];
    float msg = h[(long)src * H + c] + eav;
    msg = (msg > 0.f ? msg : 0.f) + EPSM;
    float m = msg * tl;
    atomicMax(mx + (long)dst * H + c, __float_as_uint(m));
}

// -------------------- edge pass 2: den += e^(m-mx), num += e^(m-mx)*msg ----------
__global__ __launch_bounds__(256) void k_edge_sum(const int* __restrict__ ei,
                                                  const int* __restrict__ ea,
                                                  const float* __restrict__ bemb,
                                                  const float* __restrict__ h,
                                                  const float* __restrict__ t, int l,
                                                  const unsigned* __restrict__ mx,
                                                  float* __restrict__ den,
                                                  float* __restrict__ num) {
    __shared__ float sbe[BOND_F * BOND_V * H];
    for (int i = threadIdx.x; i < BOND_F * BOND_V * H; i += 256) sbe[i] = bemb[i];
    __syncthreads();
    int e = blockIdx.x * 4 + (threadIdx.x >> 6);
    int c = threadIdx.x & 63;
    if (e >= N_EDGES) return;
    float tl = t[l];
    int src = ei[e], dst = ei[N_EDGES + e];
    float eav = sbe[ea[e * 3 + 0] * H + c]
              + sbe[(ea[e * 3 + 1] + BOND_V) * H + c]
              + sbe[(ea[e * 3 + 2] + 2 * BOND_V) * H + c];
    float msg = h[(long)src * H + c] + eav;
    msg = (msg > 0.f ? msg : 0.f) + EPSM;
    float m = msg * tl;
    float ex = __expf(m - __uint_as_float(mx[(long)dst * H + c]));
    atomicAdd(den + (long)dst * H + c, ex);
    atomicAdd(num + (long)dst * H + c, ex * msg);
}

// -------------------- o = num/(den+1e-16) + h --------------------
__global__ void k_combine(const float* __restrict__ num, const float* __restrict__ den,
                          const float* __restrict__ h, float* __restrict__ o) {
    long i = (long)blockIdx.x * blockDim.x + threadIdx.x;
    if (i >= (long)N_NODES * H) return;
    o[i] = num[i] / (den[i] + 1e-16f) + h[i];
}

// -------------------- fused MLP: [16,64]x[64,128] -> LN -> relu -> [16,128]x[128,64]
// One wave per 16-node tile; 32x v_wmma_f32_16x16x32_bf16 per wave, f32 accumulate.
// A-fragment layout (16-bit A 16x32, contiguous k, not ISA table order): lane&15 = row M,
//   K(j) = (lane>=16 ? 16 : 0) + j  (+32*kc).
// C/D layout: lane&15 = col N, VGPR v -> row M = v + 8*(lane>=16).
__global__ __launch_bounds__(128) void k_mlp(const float* __restrict__ o,
                                             float* __restrict__ cur,
                                             const unsigned short* __restrict__ p1,
                                             const unsigned short* __restrict__ p2,
                                             const float* __restrict__ bias1,
                                             const float* __restrict__ ln1g,
                                             const float* __restrict__ ln1b,
                                             const float* __restrict__ bias2,
                                             int accumulate) {
    __shared__ float mid_s[4][16 * H2];          // 32KB: one 16x128 tile per wave
    int wave = threadIdx.x >> 5;
    int lane = threadIdx.x & 31;
    int half = lane >> 4;
    int rl   = lane & 15;
    int tile = blockIdx.x * 4 + wave;
    float* mid = &mid_s[wave][0];

    int grow = tile * 16 + rl;
    if (grow > N_NODES - 1) grow = N_NODES - 1;   // clamp loads; stores guarded below
    const float* orow = o + (long)grow * H;
    int kb = half * 16;

    // A fragments for GEMM1 (K=64 -> two K=32 chunks)
    v16bf a0, a1;
#pragma unroll
    for (int j = 0; j < 16; ++j) {
        int k = kb + j;
        a0[j] = (__bf16)orow[k];
        a1[j] = (__bf16)orow[32 + k];
    }

    // GEMM1: mid[16,128] = A[16,64] x W1[64,128]
    v8f acc[8];
#pragma unroll
    for (int nt = 0; nt < 8; ++nt) {
        v8f c = {};
        v16bf b0 = *(const v16bf*)(p1 + ((nt * 2 + 0) * 32 + lane) * 16);
        v16bf b1 = *(const v16bf*)(p1 + ((nt * 2 + 1) * 32 + lane) * 16);
        c = __builtin_amdgcn_wmma_f32_16x16x32_bf16(false, a0, false, b0, (short)0, c, false, false);
        c = __builtin_amdgcn_wmma_f32_16x16x32_bf16(false, a1, false, b1, (short)0, c, false, false);
        acc[nt] = c;
    }

    // scatter D tiles (+bias1) into LDS
#pragma unroll
    for (int nt = 0; nt < 8; ++nt) {
#pragma unroll
        for (int v = 0; v < 8; ++v) {
            int col = nt * 16 + rl;
            mid[(v + half * 8) * H2 + col] = acc[nt][v] + bias1[col];
        }
    }
    __syncthreads();

    // LayerNorm(128) + relu, one row per lane (lanes 0..15)
    if (lane < 16) {
        float* r = mid + lane * H2;
        float s = 0.f, sq = 0.f;
        for (int j = 0; j < H2; ++j) { float v = r[j]; s += v; sq += v * v; }
        float mu  = s * (1.f / H2);
        float inv = rsqrtf(sq * (1.f / H2) - mu * mu + 1e-5f);
        for (int j = 0; j < H2; ++j) {
            float v = (r[j] - mu) * inv * ln1g[j] + ln1b[j];
            r[j] = v > 0.f ? v : 0.f;
        }
    }
    __syncthreads();

    // GEMM2: out[16,64] = mid[16,128] x W2[128,64]
    v8f acc2[4];
#pragma unroll
    for (int nt = 0; nt < 4; ++nt) {
        v8f c = {};
#pragma unroll
        for (int kc = 0; kc < 4; ++kc) {
            v16bf af;
#pragma unroll
            for (int j = 0; j < 16; ++j) {
                int k = kc * 32 + kb + j;
                af[j] = (__bf16)mid[rl * H2 + k];
            }
            v16bf bf = *(const v16bf*)(p2 + ((nt * 4 + kc) * 32 + lane) * 16);
            c = __builtin_amdgcn_wmma_f32_16x16x32_bf16(false, af, false, bf, (short)0, c, false, false);
        }
        acc2[nt] = c;
    }

    // writeout (+bias2, residual for layers >= 1)
#pragma unroll
    for (int nt = 0; nt < 4; ++nt) {
#pragma unroll
        for (int v = 0; v < 8; ++v) {
            int col = nt * 16 + rl;
            int row = tile * 16 + v + half * 8;
            if (row < N_NODES) {
                float val = acc2[nt][v] + bias2[col];
                long idx = (long)row * H + col;
                if (accumulate) cur[idx] += val; else cur[idx] = val;
            }
        }
    }
}

// -------------------- final divide --------------------
__global__ void k_div(const float* __restrict__ psum, const float* __restrict__ pcnt,
                      float* __restrict__ out) {
    int i = blockIdx.x * blockDim.x + threadIdx.x;
    if (i >= NGRAPH * H) return;
    float c = pcnt[i >> 6];
    out[i] = psum[i] / (c > 1.f ? c : 1.f);
}

extern "C" void kernel_launch(void* const* d_in, const int* in_sizes, int n_in,
                              void* d_out, int out_size, void* d_ws, size_t ws_size,
                              hipStream_t stream) {
    (void)in_sizes; (void)n_in; (void)out_size; (void)ws_size;
    const int*   x     = (const int*)d_in[0];
    const int*   eattr = (const int*)d_in[1];
    const int*   eidx  = (const int*)d_in[2];
    const int*   batch = (const int*)d_in[3];
    const float* aemb  = (const float*)d_in[4];
    const float* bemb  = (const float*)d_in[5];
    const float* t     = (const float*)d_in[6];
    const float* W1    = (const float*)d_in[7];
    const float* b1    = (const float*)d_in[8];
    const float* ln1g  = (const float*)d_in[9];
    const float* ln1b  = (const float*)d_in[10];
    const float* W2    = (const float*)d_in[11];
    const float* b2    = (const float*)d_in[12];
    const float* ng    = (const float*)d_in[13];
    const float* nb    = (const float*)d_in[14];

    const long NH = (long)N_NODES * H;
    // workspace layout (~143.6 MB): 7x [N,64] f32, pool, packed bf16 weights
    float* ws   = (float*)d_ws;
    float* xn   = ws;
    float* cur  = xn  + NH;
    float* hh   = cur + NH;
    unsigned* mx = (unsigned*)(hh + NH);
    float* den  = (float*)mx + NH;
    float* num  = den + NH;
    float* obuf = num + NH;
    float* psum = obuf + NH;
    float* pcnt = psum + (long)NGRAPH * H;
    unsigned short* p1 = (unsigned short*)(pcnt + NGRAPH);   // 32B-aligned by construction
    unsigned short* p2 = p1 + L_LAYERS * H * H2;

    // zero pooling accumulators
    k_zero<<<(NGRAPH * H + NGRAPH + 255) / 256, 256, 0, stream>>>(psum, NGRAPH * H + NGRAPH);
    // atom encoder
    k_atom_enc<<<(int)((NH + 255) / 256), 256, 0, stream>>>(x, aemb, xn);
    // pack weights (all layers) into bf16 WMMA B-fragment layout
    k_pack<<<(L_LAYERS * H * H2 + 255) / 256, 256, 0, stream>>>(W1, p1, H, H2);
    k_pack<<<(L_LAYERS * H * H2 + 255) / 256, 256, 0, stream>>>(W2, p2, H2, H);

    const int EB = (N_EDGES + 3) / 4;
    for (int l = 0; l < L_LAYERS; ++l) {
        const float* h = (l == 0) ? xn : hh;
        if (l > 0)
            k_ln_node<<<N_NODES, H, 0, stream>>>(cur, hh, ng + l * H, nb + l * H);
        // zero mx/den/num (contiguous)
        k_zero<<<(int)((3 * NH + 255) / 256), 256, 0, stream>>>((float*)mx, 3 * NH);
        k_edge_max<<<EB, 256, 0, stream>>>(eidx, eattr, bemb, h, t, l, mx);
        k_edge_sum<<<EB, 256, 0, stream>>>(eidx, eattr, bemb, h, t, l, mx, den, num);
        k_combine<<<(int)((NH + 255) / 256), 256, 0, stream>>>(num, den, h, obuf);
        k_mlp<<<(N_NODES / 16 + 3) / 4, 128, 0, stream>>>(
            obuf, cur, p1 + l * H * H2, p2 + l * H * H2,
            b1 + l * H2, ln1g + l * H2, ln1b + l * H2, b2 + l * H, l > 0 ? 1 : 0);
    }
    k_ln_pool<<<N_NODES, H, 0, stream>>>(cur, batch, ng, nb, psum, pcnt);
    k_div<<<(NGRAPH * H + 255) / 256, 256, 0, stream>>>(psum, pcnt, (float*)d_out);
}
